// ArgClassifierLayer_68006512164937
// MI455X (gfx1250) — matrix operand order repaired and checked by reading, not compiled, not run on hardware
//
#include <hip/hip_runtime.h>
#include <hip/hip_bf16.h>

// ---- problem dims ----
#define B_    4096
#define E_    28
#define D_    500
#define K1    1024   // padded 2*D
#define M_    500
#define MP    512    // padded M
#define H_    64
#define R_    16
#define NEXP  34
#define SPB   2      // samples per block

typedef __attribute__((ext_vector_type(16))) __bf16 v16bf;
typedef __attribute__((ext_vector_type(8)))  __bf16 v8bf;
typedef __attribute__((ext_vector_type(4)))  __bf16 v4bf;
typedef __attribute__((ext_vector_type(8)))  float  v8f;
typedef __attribute__((ext_vector_type(4)))  float  v4f;

union FragU { v16bf v; v8bf h[2]; };

__device__ __forceinline__ v8f vzero8() {
  v8f v;
  for (int i = 0; i < 8; ++i) v[i] = 0.f;
  return v;
}

// ---------------------------------------------------------------------------
// Pre-pass 1: Wm [1000,500] f32 -> WmT [512 cols][1024 K] bf16 (zero padded).
// ---------------------------------------------------------------------------
__global__ __launch_bounds__(256) void convert_wm(const float* __restrict__ Wm,
                                                  __bf16* __restrict__ wmT) {
  int idx = blockIdx.x * 256 + threadIdx.x;       // idx = n*1024 + k
  if (idx >= MP * K1) return;
  int n = idx >> 10;
  int k = idx & (K1 - 1);
  float v = (n < M_ && k < 2 * D_) ? Wm[k * M_ + n] : 0.f;
  wmT[idx] = (__bf16)v;
}

// ---------------------------------------------------------------------------
// Pre-pass 2: W1 [34,500,64] f32 -> w1T [34][64 cols][512 K] bf16 (padded).
// ---------------------------------------------------------------------------
__global__ __launch_bounds__(256) void convert_w1(const float* __restrict__ W1,
                                                  __bf16* __restrict__ w1t) {
  int idx = blockIdx.x * 256 + threadIdx.x;       // idx = e*64*512 + h*512 + m
  if (idx >= NEXP * H_ * MP) return;
  int e   = idx / (H_ * MP);
  int rem = idx - e * (H_ * MP);
  int h   = rem >> 9;
  int m   = rem & (MP - 1);
  float v = (m < M_) ? W1[(e * M_ + m) * H_ + h] : 0.f;
  w1t[idx] = (__bf16)v;
}

// ---------------------------------------------------------------------------
// Fused main kernel: TWO samples per block, 8 wave32 (256 threads).
//   stage1: merged[64,512] = relu(X[64,1024] @ WmT^T + bm)      (bf16 WMMA)
//   stage2: h[64,64]       = relu(merged @ w1T[e_s]^T + b1)     (bf16 WMMA)
//   stage3: out[2][28,16]  = h @ W2[e_s] + b2                   (bf16 WMMA)
// Rows 0..31 = sample 0 (28 valid), rows 32..63 = sample 1.
// ---------------------------------------------------------------------------
__global__ __launch_bounds__(256) void fused_moe(
    const float* __restrict__ arg,    // [B,28,500]
    const float* __restrict__ ev,     // [B,1,500]
    const int*   __restrict__ evt,    // [B]
    const float* __restrict__ bm,     // [500]
    const float* __restrict__ b1,     // [34,64]
    const float* __restrict__ b2,     // [34,16]
    const float* __restrict__ W2,     // [34,64,16]
    const __bf16* __restrict__ wmT,   // [512,1024]
    const __bf16* __restrict__ w1t,   // [34,64,512]
    float* __restrict__ out)          // [B,28,16]
{
  // Row strides: multiples of 8 bf16 (16B -> aligned b128) with bank spread.
  __shared__ __align__(16) __bf16 x_chunk[64][136];   // 17.4 KB (Kc=128)
  __shared__ __align__(16) __bf16 merged_s[64][520];  // 66.6 KB
  __shared__ __align__(16) __bf16 h_s[64][72];        //  9.2 KB
  __shared__ __align__(16) __bf16 w2t_s[SPB][R_][72]; //  4.6 KB (K-contig B tile)
  __shared__ float bm_s[MP];                          //  2.0 KB
  __shared__ float b1_s[SPB][H_];
  __shared__ float b2_s[SPB][R_];

  const int b0   = blockIdx.x * SPB;  // first sample of this block
  const int tid  = threadIdx.x;
  const int wv   = tid >> 5;          // wave 0..7
  const int lane = tid & 31;
  const int half = lane >> 4;         // 0/1: lane group
  const int ln   = lane & 15;
  const int e0   = evt[b0];
  const int e1   = evt[b0 + 1];

  // ---- preload small per-sample params into LDS ----
  for (int i = tid; i < MP; i += 256) bm_s[i] = (i < M_) ? bm[i] : 0.f;
  // W2[e] transposed to [n][k] bf16 so B-fragments are contiguous-K loads
  for (int i = tid; i < SPB * R_ * H_; i += 256) {
    const int s = i >> 10;
    const int n = (i >> 6) & 15;
    const int k = i & 63;
    const int es = s ? e1 : e0;
    w2t_s[s][n][k] = (__bf16)W2[es * (H_ * R_) + k * R_ + n];
  }
  if (tid < H_) { b1_s[0][tid] = b1[e0 * H_ + tid]; b1_s[1][tid] = b1[e1 * H_ + tid]; }
  if (tid < R_) { b2_s[0][tid] = b2[e0 * R_ + tid]; b2_s[1][tid] = b2[e1 * R_ + tid]; }

  // ================= Stage 1: merged = relu(X @ Wm + bm) =================
  // wave wv owns col-tiles 4*wv+j (j=0..3) x row-tiles rt=0..3 -> 16 tiles
  v8f acc[4][4];
#pragma unroll
  for (int rt = 0; rt < 4; ++rt)
#pragma unroll
    for (int j = 0; j < 4; ++j) acc[rt][j] = vzero8();

  const int r_ld   = tid >> 2;        // 0..63 : row of X loaded by this thread
  const int sub_ld = tid & 3;         // 0..3  : 32-element K sub-block
  const int s_ld   = r_ld >> 5;       // sample of that row
  const int rr_ld  = r_ld & 31;       // row within sample
  const bool rowok = (rr_ld < E_);

  for (int kc = 0; kc < 8; ++kc) {    // K chunks of 128
    __syncthreads();                  // protect x_chunk from previous readers
    // ---- stage X = concat(arg, ev) -> bf16 LDS (vector fast path) ----
    {
      const int kb = kc * 128 + sub_ld * 32;
      const float* src = nullptr;
      if (rowok) {
        if (kb + 32 <= D_)
          src = arg + ((size_t)(b0 + s_ld) * E_ + rr_ld) * D_ + kb;
        else if (kb >= D_ && kb + 32 <= 2 * D_)
          src = ev + (size_t)(b0 + s_ld) * D_ + (kb - D_);
      }
      if (src) {                      // whole 32-block inside one source
#pragma unroll
        for (int i = 0; i < 8; ++i) {
          const v4f f = *(const v4f*)(src + 4 * i);   // global_load_b128
          v4bf t;
          t[0] = (__bf16)f[0]; t[1] = (__bf16)f[1];
          t[2] = (__bf16)f[2]; t[3] = (__bf16)f[3];
          *(v4bf*)&x_chunk[r_ld][sub_ld * 32 + 4 * i] = t;  // ds_store_b64
        }
      } else {                        // boundary / padded rows / K pad
#pragma unroll
        for (int i = 0; i < 32; ++i) {
          const int K = kb + i;
          float v = 0.f;
          if (rowok) {
            if (K < D_)           v = arg[((size_t)(b0 + s_ld) * E_ + rr_ld) * D_ + K];
            else if (K < 2 * D_)  v = ev[(size_t)(b0 + s_ld) * D_ + (K - D_)];
          }
          x_chunk[r_ld][sub_ld * 32 + i] = (__bf16)v;
        }
      }
    }
    __syncthreads();

    // prefetch next chunk of this wave's WmT columns into L2/L0
    if (kc + 1 < 8)
      __builtin_prefetch(wmT + (size_t)(16 * (4 * wv) + ln) * K1 + (kc + 1) * 128, 0, 1);

#pragma unroll
    for (int ks = 0; ks < 4; ++ks) {  // k-steps of 32 within chunk
      const int k0 = ks * 32;
      FragU a[4];
#pragma unroll
      for (int rt = 0; rt < 4; ++rt) {
        // 16-bit A layout: lane -> row ln; VGPR0-3: K=k0+8h+0..7, VGPR4-7: +16
        a[rt].h[0] = *(const v8bf*)&x_chunk[16 * rt + ln][k0 + 8 * half];
        a[rt].h[1] = *(const v8bf*)&x_chunk[16 * rt + ln][k0 + 16 + 8 * half];
      }
#pragma unroll
      for (int j = 0; j < 4; ++j) {
        const int col = 16 * (4 * wv + j) + ln;
        // B layout: lane -> col, K=k0+8h+0..7 then k0+16+8h+0..7
        const __bf16* p = wmT + (size_t)col * K1 + (kc * 128 + k0) + 8 * half;
        FragU bb;
        bb.h[0] = *(const v8bf*)(p);
        bb.h[1] = *(const v8bf*)(p + 16);
#pragma unroll
        for (int rt = 0; rt < 4; ++rt)
          acc[rt][j] = __builtin_amdgcn_wmma_f32_16x16x32_bf16(
              false, a[rt].v, false, bb.v, (short)0, acc[rt][j], false, false);
      }
    }
  }

  // epilogue: +bias, relu, convert, stash in LDS (C layout: row = v + 8*half)
#pragma unroll
  for (int rt = 0; rt < 4; ++rt)
#pragma unroll
    for (int j = 0; j < 4; ++j) {
      const int col = 16 * (4 * wv + j) + ln;
      const float bmv = bm_s[col];
#pragma unroll
      for (int v = 0; v < 8; ++v) {
        const int row = 16 * rt + 8 * half + v;
        merged_s[row][col] = (__bf16)fmaxf(acc[rt][j][v] + bmv, 0.f);
      }
    }
  __syncthreads();

  // ================= Stage 2: h = relu(merged @ w1[e_s] + b1[e_s]) =======
  // 16 tiles: [sample s][row-tile rt 0..1][col-tile ct 0..3]; 2 per wave.
#pragma unroll
  for (int i = 0; i < 2; ++i) {
    const int T  = 2 * wv + i;
    const int s  = T >> 3;
    const int rt = (T >> 2) & 1;
    const int ct = T & 3;
    const int es = s ? e1 : e0;
    v8f acc2 = vzero8();
    const __bf16* w1b = w1t + ((size_t)es * H_ + (16 * ct + ln)) * MP;
#pragma unroll
    for (int ks = 0; ks < 16; ++ks) {
      const int k0 = ks * 32;
      FragU a, bb;
      a.h[0]  = *(const v8bf*)&merged_s[32 * s + 16 * rt + ln][k0 + 8 * half];
      a.h[1]  = *(const v8bf*)&merged_s[32 * s + 16 * rt + ln][k0 + 16 + 8 * half];
      const __bf16* p = w1b + k0 + 8 * half;
      bb.h[0] = *(const v8bf*)(p);
      bb.h[1] = *(const v8bf*)(p + 16);
      acc2 = __builtin_amdgcn_wmma_f32_16x16x32_bf16(
          false, a.v, false, bb.v, (short)0, acc2, false, false);
    }
    const int col = 16 * ct + ln;
    const float bias = b1_s[s][col];
#pragma unroll
    for (int v = 0; v < 8; ++v) {
      const int row = 32 * s + 16 * rt + 8 * half + v;
      h_s[row][col] = (__bf16)fmaxf(acc2[v] + bias, 0.f);
    }
  }
  __syncthreads();

  // ================= Stage 3: logits = h @ w2[e_s] + b2[e_s] (WMMA) ======
  // 4 tiles of 16 rows x 16 cols, K=64; waves 0..3 take one tile each.
  if (wv < 4) {
    const int rt = wv;                // row-tile over the 64 rows
    const int s  = rt >> 1;           // sample of this row-tile
    v8f acc3 = vzero8();
#pragma unroll
    for (int ks = 0; ks < 2; ++ks) {
      const int k0 = ks * 32;
      FragU a, bb;
      a.h[0]  = *(const v8bf*)&h_s[16 * rt + ln][k0 + 8 * half];
      a.h[1]  = *(const v8bf*)&h_s[16 * rt + ln][k0 + 16 + 8 * half];
      bb.h[0] = *(const v8bf*)&w2t_s[s][ln][k0 + 8 * half];
      bb.h[1] = *(const v8bf*)&w2t_s[s][ln][k0 + 16 + 8 * half];
      acc3 = __builtin_amdgcn_wmma_f32_16x16x32_bf16(
          false, a.v, false, bb.v, (short)0, acc3, false, false);
    }
    const float bias = b2_s[s][ln];
#pragma unroll
    for (int v = 0; v < 8; ++v) {
      const int row = 16 * rt + 8 * half + v;
      const int rr  = row & 31;       // row within sample
      if (rr < E_)
        out[((size_t)(b0 + s) * E_ + rr) * R_ + ln] = acc3[v] + bias;
    }
  }
}

// ---------------------------------------------------------------------------
extern "C" void kernel_launch(void* const* d_in, const int* in_sizes, int n_in,
                              void* d_out, int out_size, void* d_ws, size_t ws_size,
                              hipStream_t stream) {
  const float* arg = (const float*)d_in[0];
  const float* ev  = (const float*)d_in[1];
  const int*   evt = (const int*)d_in[2];
  const float* Wm  = (const float*)d_in[3];
  const float* bm  = (const float*)d_in[4];
  const float* W1  = (const float*)d_in[5];
  const float* b1  = (const float*)d_in[6];
  const float* W2  = (const float*)d_in[7];
  const float* b2  = (const float*)d_in[8];
  float* out = (float*)d_out;

  // workspace layout: WmT (1 MB) | w1T (2.2 MB)
  __bf16* wmT = (__bf16*)d_ws;
  __bf16* w1t = (__bf16*)((char*)d_ws + (size_t)MP * K1 * sizeof(__bf16));

  hipLaunchKernelGGL(convert_wm, dim3((MP * K1 + 255) / 256), dim3(256), 0, stream,
                     Wm, wmT);
  hipLaunchKernelGGL(convert_w1, dim3((NEXP * H_ * MP + 255) / 256), dim3(256), 0, stream,
                     W1, w1t);
  hipLaunchKernelGGL(fused_moe, dim3(B_ / SPB), dim3(256), 0, stream,
                     arg, ev, evt, bm, b1, b2, W2, wmT, w1t, out);
}
